// AbomasumLayer_67662914781944
// MI455X (gfx1250) — matrix operand order repaired and checked by reading, not compiled, not run on hardware
//
#include <hip/hip_runtime.h>
#include <math.h>

typedef __bf16 bf16_t;
typedef __attribute__((ext_vector_type(16))) __bf16 v16bf;
typedef __attribute__((ext_vector_type(8)))  __bf16 v8bf;
typedef __attribute__((ext_vector_type(8)))  float  v8f;
typedef __attribute__((ext_vector_type(4)))  int    v4i;

static constexpr int BB = 2, NN = 2048, DD = 1024, HH = 16, DHH = 64;
static constexpr int TT = BB * NN;          // 4096 tokens
static constexpr int D3 = 3 * DD;           // 3072

#define WMMA_BF16(a, b, c) \
  __builtin_amdgcn_wmma_f32_16x16x32_bf16(false, (a), false, (b), (short)0, (c), false, false)

// Async global->LDS (CDNA5): builtin confirmed present; params are
// (AS1 v4i*, AS3 v4i*, imm offset, imm cpol).
#if defined(__gfx1250__) && __has_builtin(__builtin_amdgcn_global_load_async_to_lds_b128) && \
    __has_builtin(__builtin_amdgcn_s_wait_asynccnt)
#define HAVE_ASYNC_LDS 1
typedef __attribute__((address_space(1))) v4i* as1_v4i_p;
typedef __attribute__((address_space(3))) v4i* as3_v4i_p;
__device__ __forceinline__ void async_copy16(const bf16_t* g, bf16_t* l) {
  // LDS generic address: aperture in high 32 bits, byte offset in low 32 bits.
  __builtin_amdgcn_global_load_async_to_lds_b128(
      (as1_v4i_p)(uintptr_t)g, (as3_v4i_p)(uint32_t)(uintptr_t)l, 0, 0);
}
#else
#define HAVE_ASYNC_LDS 0
#endif

// ---- fragment loaders (CDNA5 wave32 layouts, ISA 7.12.2) -------------------
// A (16xK=32, 16-bit): lane l -> row l&15; K = off..off+7 , 16+off..16+off+7 (off=(l>>4)*8)
__device__ __forceinline__ v16bf load_afrag(const bf16_t* base, int ld, int row0,
                                            int k0, int m, int off) {
  const bf16_t* p = base + (size_t)(row0 + m) * ld + k0 + off;
  v8bf lo = *(const v8bf*)p;
  v8bf hi = *(const v8bf*)(p + 16);
  v16bf a;
#pragma unroll
  for (int i = 0; i < 8; ++i) { a[i] = lo[i]; a[8 + i] = hi[i]; }
  return a;
}
// B (K=32 x 16, 16-bit): lane l -> col l&15; K = off..off+7 , 16+off..16+off+7 (off=(l>>4)*8)
__device__ __forceinline__ v16bf load_bfrag(const bf16_t* colbase, int ld, int col0,
                                            int k0, int m, int hseg) {
  const bf16_t* p = colbase + (size_t)(col0 + m) * ld + k0 + hseg * 8;
  v8bf lo = *(const v8bf*)p;
  v8bf hi = *(const v8bf*)(p + 16);
  v16bf b;
#pragma unroll
  for (int i = 0; i < 8; ++i) { b[i] = lo[i]; b[8 + i] = hi[i]; }
  return b;
}

// ---- stage 0: per-batch max(uncertainty) -> log-confidence -----------------
__global__ __launch_bounds__(256) void logconf_kernel(const float* __restrict__ u,
                                                      float* __restrict__ lc) {
  __shared__ float sred[256];
  const int b = blockIdx.x;
  const float* ub = u + b * NN;
  float mx = -1e30f;
  for (int i = threadIdx.x; i < NN; i += 256) mx = fmaxf(mx, ub[i]);
  sred[threadIdx.x] = mx;
  __syncthreads();
  for (int s = 128; s > 0; s >>= 1) {
    if (threadIdx.x < s) sred[threadIdx.x] = fmaxf(sred[threadIdx.x], sred[threadIdx.x + s]);
    __syncthreads();
  }
  const float um = sred[0] + 1e-8f;
  for (int i = threadIdx.x; i < NN; i += 256)
    lc[b * NN + i] = __logf(1.0f - ub[i] / um + 1e-8f);
}

// ---- stage 1: f32 -> bf16 convert ------------------------------------------
__global__ __launch_bounds__(256) void tobf16_kernel(const float* __restrict__ src,
                                                     bf16_t* __restrict__ dst, int n) {
  for (int i = blockIdx.x * blockDim.x + threadIdx.x; i < n; i += gridDim.x * blockDim.x)
    dst[i] = (bf16_t)src[i];
}

// ---- stage 2: QKV GEMM: C[tok, e] = X[tok, d] * Wqkv[e, d] -----------------
// wave computes a 32x64 tile (two A frags share each B frag);
// scatters into Q (scaled by 1/8), K, and V^T.
__global__ __launch_bounds__(128) void qkv_gemm_kernel(const bf16_t* __restrict__ X,
                                                       const bf16_t* __restrict__ W,
                                                       bf16_t* __restrict__ Q,
                                                       bf16_t* __restrict__ K,
                                                       bf16_t* __restrict__ Vt) {
  const int lane = threadIdx.x & 31, wid = threadIdx.x >> 5;
  const int m = lane & 15, hseg = lane >> 4, off = hseg * 8;
  const int row0 = blockIdx.x * 32;              // 128 row tiles of 32
  const int colGrp = blockIdx.y * 4 + wid;       // 48 groups of 64 features
  v8f acc0[4] = {}, acc1[4] = {};
  for (int k0 = 0; k0 < DD; k0 += 32) {
    v16bf a0 = load_afrag(X, DD, row0, k0, m, off);
    v16bf a1 = load_afrag(X, DD, row0 + 16, k0, m, off);
    if (k0 + 64 < DD) {
      __builtin_prefetch(X + (size_t)(row0 + m) * DD + k0 + 64, 0, 3);
      __builtin_prefetch(W + (size_t)(colGrp * 64 + m) * DD + k0 + 64, 0, 3);
    }
#pragma unroll
    for (int t = 0; t < 4; ++t) {
      v16bf bfr = load_bfrag(W, DD, colGrp * 64 + t * 16, k0, m, hseg);
      acc0[t] = WMMA_BF16(a0, bfr, acc0[t]);
      acc1[t] = WMMA_BF16(a1, bfr, acc1[t]);
    }
  }
#pragma unroll
  for (int t = 0; t < 4; ++t) {
    const int e = colGrp * 64 + t * 16 + m;      // feature index (wave-uniform which/h)
    const int which = e >> 10, e1 = e & 1023, h = e1 >> 6, dh = e1 & 63;
#pragma unroll
    for (int half = 0; half < 2; ++half) {
#pragma unroll
      for (int r = 0; r < 8; ++r) {
        const int tok = row0 + half * 16 + r + 8 * hseg;
        const int b = tok >> 11, n = tok & 2047;
        const float v = half ? acc1[t][r] : acc0[t][r];
        if (which == 0)
          Q[((size_t)((b * HH + h) * NN) + n) * DHH + dh] = (bf16_t)(v * 0.125f);
        else if (which == 1)
          K[((size_t)((b * HH + h) * NN) + n) * DHH + dh] = (bf16_t)v;
        else
          Vt[((size_t)((b * HH + h) * DHH) + dh) * NN + n] = (bf16_t)v;
      }
    }
  }
}

// ---- stage 3: flash attention ----------------------------------------------
// Block = 4 waves covering 4 query tiles of the SAME (b,h); the 32-key K tile
// and 64x32 V^T tile are staged in LDS once per step (async-to-LDS when
// available) and shared by all 4 waves.
__global__ __launch_bounds__(128) void attn_kernel(const bf16_t* __restrict__ Q,
                                                   const bf16_t* __restrict__ K,
                                                   const bf16_t* __restrict__ Vt,
                                                   const float* __restrict__ lcAll,
                                                   bf16_t* __restrict__ O) {
  __shared__ __align__(16) bf16_t Kt[32 * DHH];      // (key, dh)   4KB
  __shared__ __align__(16) bf16_t Vts[DHH * 32];     // (dh, key)   4KB
  __shared__ __align__(16) bf16_t Pb[4][16 * 32];    // per-wave P  4KB
  const int tid = threadIdx.x;
  const int lane = tid & 31, wid = tid >> 5;
  const int m = lane & 15, hseg = lane >> 4, off = hseg * 8;
  const int b = blockIdx.x >> 9;                     // 1024 blocks total
  const int h = (blockIdx.x >> 5) & 15;
  const int quad = blockIdx.x & 31;
  const int row0 = (quad * 4 + wid) * 16;
  const bf16_t* Qh = Q + (size_t)(b * HH + h) * NN * DHH;
  const bf16_t* Kh = K + (size_t)(b * HH + h) * NN * DHH;
  const bf16_t* Vh = Vt + (size_t)(b * HH + h) * DHH * NN;
  const float* lc = lcAll + b * NN;
  bf16_t* P = Pb[wid];

  const v16bf aq0 = load_afrag(Qh, DHH, row0, 0, m, off);   // Q tile stays in regs
  const v16bf aq1 = load_afrag(Qh, DHH, row0, 32, m, off);

  float mi[8], li[8];
  v8f o[4] = {};
#pragma unroll
  for (int r = 0; r < 8; ++r) { mi[r] = -1e30f; li[r] = 0.0f; }

  for (int j0 = 0; j0 < NN; j0 += 32) {
    // ---- stage K (contiguous 4KB) and V^T (64 rows x 32) tiles into LDS ----
#if HAVE_ASYNC_LDS
#pragma unroll
    for (int c = tid; c < 256; c += 128)
      async_copy16(Kh + (size_t)j0 * DHH + c * 8, Kt + c * 8);
#pragma unroll
    for (int c = tid; c < 256; c += 128)
      async_copy16(Vh + (size_t)(c >> 2) * NN + j0 + (c & 3) * 8, Vts + c * 8);
    __builtin_amdgcn_s_wait_asynccnt(0);
#else
#pragma unroll
    for (int c = tid; c < 256; c += 128) {
      *(v8bf*)(Kt + c * 8) = *(const v8bf*)(Kh + (size_t)j0 * DHH + c * 8);
      *(v8bf*)(Vts + c * 8) = *(const v8bf*)(Vh + (size_t)(c >> 2) * NN + j0 + (c & 3) * 8);
    }
#endif
    __syncthreads();

    v8f c0 = {}, c1 = {};
    c0 = WMMA_BF16(aq0, load_bfrag(Kt, DHH, 0, 0, m, hseg), c0);
    c0 = WMMA_BF16(aq1, load_bfrag(Kt, DHH, 0, 32, m, hseg), c0);
    c1 = WMMA_BF16(aq0, load_bfrag(Kt, DHH, 16, 0, m, hseg), c1);
    c1 = WMMA_BF16(aq1, load_bfrag(Kt, DHH, 16, 32, m, hseg), c1);

    const float bias0 = lc[j0 + m], bias1 = lc[j0 + 16 + m];
    float sc[8];
#pragma unroll
    for (int r = 0; r < 8; ++r) {
      const float s0 = c0[r] + bias0;
      const float s1 = c1[r] + bias1;
      float t = fmaxf(s0, s1);
#pragma unroll
      for (int msk = 1; msk < 16; msk <<= 1) t = fmaxf(t, __shfl_xor(t, msk, 32));
      const float mnew = fmaxf(mi[r], t);
      sc[r] = __expf(mi[r] - mnew);
      const float p0 = __expf(s0 - mnew);
      const float p1 = __expf(s1 - mnew);
      P[(r + 8 * hseg) * 32 + m] = (bf16_t)p0;        // C-layout -> A-layout transpose
      P[(r + 8 * hseg) * 32 + 16 + m] = (bf16_t)p1;   // via wave-private LDS
      float rs = p0 + p1;
#pragma unroll
      for (int msk = 1; msk < 16; msk <<= 1) rs += __shfl_xor(rs, msk, 32);
      li[r] = li[r] * sc[r] + rs;
      mi[r] = mnew;
    }
#pragma unroll
    for (int t = 0; t < 4; ++t)
#pragma unroll
      for (int r = 0; r < 8; ++r) o[t][r] *= sc[r];

    // read P back as A fragment (LDS ops in-order within a wave)
    v8bf plo = *(const v8bf*)(P + m * 32 + off);
    v8bf phi = *(const v8bf*)(P + m * 32 + 16 + off);
    v16bf pa;
#pragma unroll
    for (int i = 0; i < 8; ++i) { pa[i] = plo[i]; pa[8 + i] = phi[i]; }

#pragma unroll
    for (int t = 0; t < 4; ++t) {
      v16bf bv = load_bfrag(Vts, 32, t * 16, 0, m, hseg);   // V^T tile in LDS
      o[t] = WMMA_BF16(pa, bv, o[t]);
    }
    __syncthreads();   // protect K/V tiles before next-step overwrite
  }
#pragma unroll
  for (int t = 0; t < 4; ++t)
#pragma unroll
    for (int r = 0; r < 8; ++r) {
      const size_t tok = (size_t)(b * NN + row0 + r + 8 * hseg);
      O[tok * DD + h * DHH + t * 16 + m] = (bf16_t)(o[t][r] / li[r]);
    }
}

// ---- stage 4: out-proj + residual: Y[tok,e] = x + O[tok,d]*Wout[e,d] -------
__global__ __launch_bounds__(128) void outproj_kernel(const bf16_t* __restrict__ O,
                                                      const bf16_t* __restrict__ W,
                                                      const float* __restrict__ x,
                                                      float* __restrict__ Y) {
  const int lane = threadIdx.x & 31, wid = threadIdx.x >> 5;
  const int m = lane & 15, hseg = lane >> 4, off = hseg * 8;
  const int row0 = blockIdx.x * 32;
  const int colGrp = blockIdx.y * 4 + wid;       // 16 groups of 64
  v8f acc0[4] = {}, acc1[4] = {};
  for (int k0 = 0; k0 < DD; k0 += 32) {
    v16bf a0 = load_afrag(O, DD, row0, k0, m, off);
    v16bf a1 = load_afrag(O, DD, row0 + 16, k0, m, off);
    if (k0 + 64 < DD) {
      __builtin_prefetch(O + (size_t)(row0 + m) * DD + k0 + 64, 0, 3);
      __builtin_prefetch(W + (size_t)(colGrp * 64 + m) * DD + k0 + 64, 0, 3);
    }
#pragma unroll
    for (int t = 0; t < 4; ++t) {
      v16bf bfr = load_bfrag(W, DD, colGrp * 64 + t * 16, k0, m, hseg);
      acc0[t] = WMMA_BF16(a0, bfr, acc0[t]);
      acc1[t] = WMMA_BF16(a1, bfr, acc1[t]);
    }
  }
#pragma unroll
  for (int t = 0; t < 4; ++t) {
    const int e = colGrp * 64 + t * 16 + m;
#pragma unroll
    for (int half = 0; half < 2; ++half) {
#pragma unroll
      for (int r = 0; r < 8; ++r) {
        const size_t tok = (size_t)(row0 + half * 16 + r + 8 * hseg);
        const float v = half ? acc1[t][r] : acc0[t][r];
        Y[tok * DD + e] = x[tok * DD + e] + v;
      }
    }
  }
}

// ---- stage 5: LayerNorm ----------------------------------------------------
__global__ __launch_bounds__(256) void lnorm_kernel(const float* __restrict__ Y,
                                                    const float* __restrict__ gamma,
                                                    const float* __restrict__ beta,
                                                    float* __restrict__ out) {
  __shared__ float ssum[256], ssq[256];
  const size_t row = blockIdx.x;
  const float* y = Y + row * DD;
  float s = 0.f, sq = 0.f;
  for (int i = threadIdx.x; i < DD; i += 256) { const float v = y[i]; s += v; sq += v * v; }
  ssum[threadIdx.x] = s; ssq[threadIdx.x] = sq;
  __syncthreads();
  for (int st = 128; st > 0; st >>= 1) {
    if (threadIdx.x < st) {
      ssum[threadIdx.x] += ssum[threadIdx.x + st];
      ssq[threadIdx.x] += ssq[threadIdx.x + st];
    }
    __syncthreads();
  }
  const float mean = ssum[0] / DD;
  const float var = ssq[0] / DD - mean * mean;
  const float rstd = rsqrtf(var + 1e-5f);
  for (int i = threadIdx.x; i < DD; i += 256)
    out[row * DD + i] = (y[i] - mean) * rstd * gamma[i] + beta[i];
}

// ---- host launcher ---------------------------------------------------------
extern "C" void kernel_launch(void* const* d_in, const int* in_sizes, int n_in,
                              void* d_out, int out_size, void* d_ws, size_t ws_size,
                              hipStream_t stream) {
  const float* x     = (const float*)d_in[0];   // (B,N,D)
  const float* unc   = (const float*)d_in[1];   // (B,N)
  const float* Wqkv  = (const float*)d_in[2];   // (3D,D)
  const float* Wout  = (const float*)d_in[3];   // (D,D)
  const float* gamma = (const float*)d_in[4];
  const float* beta  = (const float*)d_in[5];
  float* out = (float*)d_out;

  char* ws = (char*)d_ws;
  size_t o = 0;
  auto take = [&](size_t bytes) { char* p = ws + o; o += (bytes + 255) & ~(size_t)255; return p; };
  bf16_t* Xbf   = (bf16_t*)take((size_t)TT * DD * 2);
  bf16_t* Wqbf  = (bf16_t*)take((size_t)D3 * DD * 2);
  bf16_t* Wobf  = (bf16_t*)take((size_t)DD * DD * 2);
  bf16_t* Qbf   = (bf16_t*)take((size_t)TT * DD * 2);
  bf16_t* Kbf   = (bf16_t*)take((size_t)TT * DD * 2);
  bf16_t* Vtbf  = (bf16_t*)take((size_t)TT * DD * 2);
  bf16_t* Obf   = (bf16_t*)take((size_t)TT * DD * 2);
  float*  Yf    = (float*)take((size_t)TT * DD * 4);
  float*  lconf = (float*)take((size_t)TT * 4);

  logconf_kernel<<<BB, 256, 0, stream>>>(unc, lconf);
  tobf16_kernel<<<2048, 256, 0, stream>>>(x, Xbf, TT * DD);
  tobf16_kernel<<<2048, 256, 0, stream>>>(Wqkv, Wqbf, D3 * DD);
  tobf16_kernel<<<1024, 256, 0, stream>>>(Wout, Wobf, DD * DD);
  qkv_gemm_kernel<<<dim3(TT / 32, 12), 128, 0, stream>>>(Xbf, Wqbf, Qbf, Kbf, Vtbf);
  attn_kernel<<<BB * HH * (NN / 64), 128, 0, stream>>>(Qbf, Kbf, Vtbf, lconf, Obf);
  outproj_kernel<<<dim3(TT / 32, 4), 128, 0, stream>>>(Obf, Wobf, x, Yf);
  lnorm_kernel<<<TT, 256, 0, stream>>>(Yf, gamma, beta, out);
}
